// SpatialMemoryModule_73375221285179
// MI455X (gfx1250) — matrix operand order repaired and checked by reading, not compiled, not run on hardware
//
#include <hip/hip_runtime.h>
#include <hip/hip_bf16.h>

// ---------------------------------------------------------------------------
// Shapes (fixed by the reference):
//   x: (32,512,32,32) -> input_flat 16384 x 1024
//   memory: 512 x 1024, w1: 512 x 1024, b1: 512, w2: 1024 x 512, b2: 1024
// Outputs (flat, concatenated): output 16.7M | att_spatial 16.7M |
//   input_flat 16.7M | memory_norm 512K | att 8.4M  floats
// ---------------------------------------------------------------------------

typedef __attribute__((ext_vector_type(16))) __bf16 v16bf;
typedef __attribute__((ext_vector_type(8)))  float  v8f;

union BFrag {
    v16bf v;
    uint4 q[2];
    unsigned short s[16];
};

static __device__ __forceinline__ unsigned short f2bf_bits(float f) {
    union { float f; unsigned int u; } v; v.f = f;
    unsigned int u = v.u;
    u += 0x7FFFu + ((u >> 16) & 1u);   // round-to-nearest-even
    return (unsigned short)(u >> 16);
}

// ---------------------------------------------------------------------------
// Row L2-norm of x (16384 rows x 1024); also copies x into the input_flat
// output section and stores per-row 1/max(||row||,eps).
// ---------------------------------------------------------------------------
__global__ void __launch_bounds__(256) k_rownorm_x(const float* __restrict__ x,
                                                   float* __restrict__ inflat_out,
                                                   float* __restrict__ invn) {
    __shared__ float red[8];
    const int row = blockIdx.x;
    const int tid = threadIdx.x;
    const float* xr = x + (size_t)row * 1024;
    float* orow = inflat_out + (size_t)row * 1024;
    float ss = 0.f;
    for (int i = tid; i < 1024; i += 256) {
        float v = xr[i];
        orow[i] = v;
        ss += v * v;
    }
    #pragma unroll
    for (int m = 1; m < 32; m <<= 1) ss += __shfl_xor(ss, m, 32);
    if ((tid & 31) == 0) red[tid >> 5] = ss;
    __syncthreads();
    if (tid == 0) {
        float t = 0.f;
        #pragma unroll
        for (int i = 0; i < 8; i++) t += red[i];
        invn[row] = 1.0f / fmaxf(sqrtf(t), 1e-12f);
    }
}

// ---------------------------------------------------------------------------
// fp32 -> bf16 bulk convert
// ---------------------------------------------------------------------------
__global__ void __launch_bounds__(256) k_cvt(const float* __restrict__ src,
                                             unsigned short* __restrict__ dst, int n) {
    int i = blockIdx.x * 256 + threadIdx.x;
    if (i < n) dst[i] = f2bf_bits(src[i]);
}

__global__ void __launch_bounds__(256) k_zero(float* __restrict__ p, int n) {
    int i = blockIdx.x * 256 + threadIdx.x;
    if (i < n) p[i] = 0.f;
}

// ---------------------------------------------------------------------------
// Generic WMMA GEMM:  C = act(A @ W^T + bias)
//   A: M x K (bf16 row-major, or f32 row-major if A_F32)
//   W: N x K bf16 row-major (so W row j supplies B column j)
//   C: M x N (f32 or bf16)
// One 16x16 output tile per wave, K-step 32 via v_wmma_f32_16x16x32_bf16.
// ---------------------------------------------------------------------------
template <bool A_F32, bool OUT_BF16, bool RELU>
__global__ void __launch_bounds__(256) k_gemm(const void* __restrict__ Ap,
                                              const unsigned short* __restrict__ W,
                                              const float* __restrict__ bias,
                                              void* __restrict__ Cp,
                                              int M, int N, int K,
                                              int lda, int ldw, int ldc) {
    const int wave   = blockIdx.x * (blockDim.x >> 5) + (threadIdx.x >> 5);
    const int tilesN = N >> 4;
    const int tiles  = (M >> 4) * tilesN;
    if (wave >= tiles) return;                       // wave-uniform: EXEC stays full
    const int tm = wave / tilesN;
    const int tn = wave - tm * tilesN;
    const int lane = threadIdx.x & 31;
    const int r  = lane & 15;
    const int hi = lane >> 4;
    const int arow = (tm << 4) + r;                  // A row this lane loads
    const int wrow = (tn << 4) + r;                  // W row (= output column)

    v8f acc = {};
    const float*          Af = (const float*)Ap;
    const unsigned short* Ab = (const unsigned short*)Ap;

    for (int kk = 0; kk < K; kk += 32) {
        BFrag a, b;
        if constexpr (A_F32) {
            const float* base = Af + (size_t)arow * lda + kk;
            #pragma unroll
            for (int e = 0; e < 8; e++) a.s[e]     = f2bf_bits(base[hi * 8 + e]);
            #pragma unroll
            for (int e = 0; e < 8; e++) a.s[8 + e] = f2bf_bits(base[16 + hi * 8 + e]);
        } else {
            const unsigned short* base = Ab + (size_t)arow * lda + kk;
            a.q[0] = *(const uint4*)(base + hi * 8);
            a.q[1] = *(const uint4*)(base + 16 + hi * 8);
        }
        const unsigned short* wb = W + (size_t)wrow * ldw + kk + hi * 8;
        b.q[0] = *(const uint4*)(wb);
        b.q[1] = *(const uint4*)(wb + 16);
        acc = __builtin_amdgcn_wmma_f32_16x16x32_bf16(false, a.v, false, b.v,
                                                      (short)0, acc, false, false);
    }

    const int col = (tn << 4) + r;
    const float bv = bias ? bias[col] : 0.f;
    #pragma unroll
    for (int e = 0; e < 8; e++) {
        const int rowo = (tm << 4) + hi * 8 + e;     // C layout: M = e + hi*8, N = lane&15
        float v = acc[e] + bv;
        if constexpr (RELU) v = fmaxf(v, 0.f);
        if constexpr (OUT_BF16)
            ((unsigned short*)Cp)[(size_t)rowo * ldc + col] = f2bf_bits(v);
        else
            ((float*)Cp)[(size_t)rowo * ldc + col] = v;
    }
}

// ---------------------------------------------------------------------------
// Row L2-norm of mem_proc (512 x 1024) -> memory_norm (f32 out) + bf16 copy
// + bf16 transposed copy (for output GEMM's B operand).
// ---------------------------------------------------------------------------
__global__ void __launch_bounds__(256) k_norm_mem(const float* __restrict__ memproc,
                                                  float* __restrict__ mn_f32,
                                                  unsigned short* __restrict__ mnbf,
                                                  unsigned short* __restrict__ mnTbf) {
    __shared__ float red[8];
    __shared__ float s_inv;
    const int row = blockIdx.x;
    const int tid = threadIdx.x;
    const float* rp = memproc + (size_t)row * 1024;
    float ss = 0.f;
    for (int i = tid; i < 1024; i += 256) { float v = rp[i]; ss += v * v; }
    #pragma unroll
    for (int m = 1; m < 32; m <<= 1) ss += __shfl_xor(ss, m, 32);
    if ((tid & 31) == 0) red[tid >> 5] = ss;
    __syncthreads();
    if (tid == 0) {
        float t = 0.f;
        #pragma unroll
        for (int i = 0; i < 8; i++) t += red[i];
        s_inv = 1.0f / fmaxf(sqrtf(t), 1e-12f);
    }
    __syncthreads();
    const float inv = s_inv;
    for (int i = tid; i < 1024; i += 256) {
        float v = rp[i] * inv;
        mn_f32[(size_t)row * 1024 + i] = v;
        unsigned short us = f2bf_bits(v);
        mnbf[(size_t)row * 1024 + i] = us;
        mnTbf[(size_t)i * 512 + row] = us;           // transpose: W[j,k]=memory_norm[k,j]
    }
}

// ---------------------------------------------------------------------------
// Cross-wave row reduction for the fused attention kernel.
// p[e] holds the lane's partial for strip-row (hi*8 + e); after the call,
// full[e] holds the reduction over all 512 columns for that row.
// ---------------------------------------------------------------------------
template <bool IS_MAX>
static __device__ __forceinline__ void row_reduce(float p[8], float full[8],
                                                  float (&lds)[16][8],
                                                  int lane, int w, int hi) {
    #pragma unroll
    for (int m = 1; m <= 8; m <<= 1) {
        #pragma unroll
        for (int e = 0; e < 8; e++) {
            float o = __shfl_xor(p[e], m, 32);
            p[e] = IS_MAX ? fmaxf(p[e], o) : (p[e] + o);
        }
    }
    if ((lane & 15) == 0) {
        #pragma unroll
        for (int e = 0; e < 8; e++) lds[hi * 8 + e][w] = p[e];
    }
    __syncthreads();
    #pragma unroll
    for (int e = 0; e < 8; e++) {
        float v = lds[hi * 8 + e][0];
        #pragma unroll
        for (int j = 1; j < 8; j++) {
            float o = lds[hi * 8 + e][j];
            v = IS_MAX ? fmaxf(v, o) : (v + o);
        }
        full[e] = v;
    }
    __syncthreads();
}

// ---------------------------------------------------------------------------
// Fused attention: one block computes a 16 x 512 score strip
//   scores = (x * invn) @ memory_norm^T          (WMMA bf16, f32 accum)
//   softmax(row) -> hardshrink(0.0025) -> L1 renorm
// writes att (f32) and accumulates per-image channel sums (atomicAdd).
// 8 waves/block; wave w owns columns [w*64, w*64+64).
// ---------------------------------------------------------------------------
__global__ void __launch_bounds__(256) k_attn(const float* __restrict__ x,
                                              const float* __restrict__ invn,
                                              const unsigned short* __restrict__ mnbf,
                                              float* __restrict__ att_out,
                                              float* __restrict__ att_avg) {
    __shared__ float lds[16][8];
    const int tid  = threadIdx.x;
    const int w    = tid >> 5;
    const int lane = tid & 31;
    const int r    = lane & 15;
    const int hi   = lane >> 4;
    const int m0   = blockIdx.x << 4;

    const int arow = m0 + r;
    const float sc = invn[arow];
    const float* abase = x + (size_t)arow * 1024;

    v8f zero = {};
    v8f acc[4] = { zero, zero, zero, zero };

    for (int kk = 0; kk < 1024; kk += 32) {
        BFrag a;
        #pragma unroll
        for (int e = 0; e < 8; e++) a.s[e]     = f2bf_bits(abase[kk + hi * 8 + e] * sc);
        #pragma unroll
        for (int e = 0; e < 8; e++) a.s[8 + e] = f2bf_bits(abase[kk + 16 + hi * 8 + e] * sc);
        #pragma unroll
        for (int t = 0; t < 4; t++) {
            const int wrow = (w << 6) + (t << 4) + r;
            const unsigned short* wb = mnbf + (size_t)wrow * 1024 + kk + hi * 8;
            BFrag b;
            b.q[0] = *(const uint4*)(wb);
            b.q[1] = *(const uint4*)(wb + 16);
            acc[t] = __builtin_amdgcn_wmma_f32_16x16x32_bf16(false, a.v, false, b.v,
                                                             (short)0, acc[t], false, false);
        }
    }

    float part[8], full[8];

    // row max
    #pragma unroll
    for (int e = 0; e < 8; e++) {
        float m = acc[0][e];
        #pragma unroll
        for (int t = 1; t < 4; t++) m = fmaxf(m, acc[t][e]);
        part[e] = m;
    }
    row_reduce<true>(part, full, lds, lane, w, hi);

    // exp
    #pragma unroll
    for (int t = 0; t < 4; t++)
        #pragma unroll
        for (int e = 0; e < 8; e++) acc[t][e] = __expf(acc[t][e] - full[e]);

    // row sum -> softmax
    #pragma unroll
    for (int e = 0; e < 8; e++) {
        float s = 0.f;
        #pragma unroll
        for (int t = 0; t < 4; t++) s += acc[t][e];
        part[e] = s;
    }
    row_reduce<false>(part, full, lds, lane, w, hi);
    #pragma unroll
    for (int t = 0; t < 4; t++)
        #pragma unroll
        for (int e = 0; e < 8; e++) acc[t][e] = acc[t][e] / full[e];

    // hard shrink
    #pragma unroll
    for (int t = 0; t < 4; t++)
        #pragma unroll
        for (int e = 0; e < 8; e++) acc[t][e] = fmaxf(acc[t][e] - 0.0025f, 0.f);

    // L1 renorm
    #pragma unroll
    for (int e = 0; e < 8; e++) {
        float s = 0.f;
        #pragma unroll
        for (int t = 0; t < 4; t++) s += acc[t][e];
        part[e] = s;
    }
    row_reduce<false>(part, full, lds, lane, w, hi);
    #pragma unroll
    for (int e = 0; e < 8; e++) full[e] = 1.0f / fmaxf(full[e], 1e-12f);
    #pragma unroll
    for (int t = 0; t < 4; t++)
        #pragma unroll
        for (int e = 0; e < 8; e++) acc[t][e] *= full[e];

    // store att (f32)
    #pragma unroll
    for (int t = 0; t < 4; t++) {
        const int col = (w << 6) + (t << 4) + r;
        #pragma unroll
        for (int e = 0; e < 8; e++) {
            const int row = m0 + hi * 8 + e;
            att_out[(size_t)row * 512 + col] = acc[t][e];
        }
    }

    // per-image channel sums (divide by 512 happens in broadcast kernel)
    const int nimg = m0 >> 9;                         // 512 rows per image
    #pragma unroll
    for (int t = 0; t < 4; t++) {
        float cs = 0.f;
        #pragma unroll
        for (int e = 0; e < 8; e++) cs += acc[t][e];
        cs += __shfl_xor(cs, 16, 32);                 // combine both row halves
        if (hi == 0) {
            const int col = (w << 6) + (t << 4) + r;
            atomicAdd(&att_avg[(size_t)nimg * 512 + col], cs);
        }
    }
}

// ---------------------------------------------------------------------------
// att_spatial[n,m,h,w] = att_channel_sum[n,m] / 512
// ---------------------------------------------------------------------------
__global__ void __launch_bounds__(256) k_bcast(const float* __restrict__ avg,
                                               float* __restrict__ out) {
    size_t idx = (size_t)blockIdx.x * 256 + threadIdx.x;   // 16,777,216 total
    out[idx] = avg[idx >> 10] * (1.0f / 512.0f);
}

// ---------------------------------------------------------------------------
extern "C" void kernel_launch(void* const* d_in, const int* in_sizes, int n_in,
                              void* d_out, int out_size, void* d_ws, size_t ws_size,
                              hipStream_t stream) {
    const float* x      = (const float*)d_in[0];   // 16384 x 1024
    const float* memory = (const float*)d_in[1];   // 512 x 1024
    const float* w1     = (const float*)d_in[2];   // 512 x 1024
    const float* b1     = (const float*)d_in[3];   // 512
    const float* w2     = (const float*)d_in[4];   // 1024 x 512
    const float* b2     = (const float*)d_in[5];   // 1024

    float* out        = (float*)d_out;
    float* out_output = out;                        // 16,777,216
    float* out_attsp  = out + 16777216;             // 16,777,216
    float* out_inflat = out + 33554432;             // 16,777,216
    float* out_mnorm  = out + 50331648;             //    524,288
    float* out_att    = out + 50855936;             //  8,388,608

    char* ws = (char*)d_ws;                         // ~8 MB used, all 16B-aligned
    float*          invn    = (float*)         (ws + 0);         //  65,536 B
    unsigned short* mem_bf  = (unsigned short*)(ws + 65536);     // 1,048,576
    unsigned short* w1_bf   = (unsigned short*)(ws + 1114112);   // 1,048,576
    unsigned short* w2_bf   = (unsigned short*)(ws + 2162688);   // 1,048,576
    unsigned short* h_bf    = (unsigned short*)(ws + 3211264);   //   524,288
    float*          memproc = (float*)         (ws + 3735552);   // 2,097,152
    unsigned short* mn_bf   = (unsigned short*)(ws + 5832704);   // 1,048,576
    unsigned short* mnT_bf  = (unsigned short*)(ws + 6881280);   // 1,048,576
    float*          att_avg = (float*)         (ws + 7929856);   //    65,536

    // 1. input row norms + input_flat copy
    k_rownorm_x<<<16384, 256, 0, stream>>>(x, out_inflat, invn);

    // 2. bf16 conversions of the static operands
    k_cvt<<<2048, 256, 0, stream>>>(memory, mem_bf, 524288);
    k_cvt<<<2048, 256, 0, stream>>>(w1, w1_bf, 524288);
    k_cvt<<<2048, 256, 0, stream>>>(w2, w2_bf, 524288);
    k_zero<<<64, 256, 0, stream>>>(att_avg, 16384);

    // 3. memory MLP (WMMA):  h = relu(memory @ w1^T + b1)   (512x512, bf16 out)
    k_gemm<false, true, true><<<128, 256, 0, stream>>>(
        mem_bf, w1_bf, b1, h_bf, 512, 512, 1024, 1024, 1024, 512);

    // 4. mem_proc = relu(h @ w2^T + b2)   (512x1024, f32 out)
    k_gemm<false, false, true><<<256, 256, 0, stream>>>(
        h_bf, w2_bf, b2, memproc, 512, 1024, 512, 512, 512, 1024);

    // 5. memory_norm (f32 out + bf16 + bf16-transposed)
    k_norm_mem<<<512, 256, 0, stream>>>(memproc, out_mnorm, mn_bf, mnT_bf);

    // 6. fused attention: scores (WMMA) -> softmax -> shrink -> L1 renorm
    k_attn<<<1024, 256, 0, stream>>>(x, invn, mn_bf, out_att, att_avg);

    // 7. output = att @ memory_norm   (WMMA, A = f32 att converted on the fly)
    k_gemm<true, false, false><<<8192, 256, 0, stream>>>(
        out_att, mnT_bf, nullptr, out_output, 16384, 1024, 512, 512, 512, 1024);

    // 8. att_spatial broadcast
    k_bcast<<<65536, 256, 0, stream>>>(att_avg, out_attsp);
}
